// Block_12463995093095
// MI455X (gfx1250) — matrix operand order repaired and checked by reading, not compiled, not run on hardware
//
#include <hip/hip_runtime.h>
#include <hip/hip_bf16.h>
#include <math.h>

// ---------------------------------------------------------------------------
// Problem constants (reference: B=1, H=W=96, C=128, HEADS=4, K=7)
// ---------------------------------------------------------------------------
#define IMG_H   96
#define IMG_W   96
#define NTOK    (IMG_H * IMG_W)      // 9216 tokens
#define CDIM    128
#define NHEADS  4
#define HDIM    32                   // CDIM / NHEADS
#define KWIN    7

typedef __attribute__((ext_vector_type(16))) _Float16 v16h;
typedef __attribute__((ext_vector_type(8)))  _Float16 v8h;
typedef __attribute__((ext_vector_type(8)))  float    v8f;

// 32-bit LDS offset of a shared-memory pointer (for async-to-LDS inline asm)
__device__ __forceinline__ unsigned lds_offset(const void* p) {
  return (unsigned)(unsigned long long)
      (__attribute__((address_space(3))) const void*)p;
}

// ---------------------------------------------------------------------------
// f32 -> f16 conversion (x and weights into workspace)
// ---------------------------------------------------------------------------
__global__ void cvt_f32_to_f16(const float* __restrict__ in,
                               _Float16* __restrict__ out, int n) {
  int i = blockIdx.x * blockDim.x + threadIdx.x;
  int stride = gridDim.x * blockDim.x;
  for (; i < n; i += stride) out[i] = (_Float16)in[i];
}

// ---------------------------------------------------------------------------
// WMMA GEMM v3:  D[M,N] = A[M,K] * W[N,K]^T + bias  (optionally exact GELU)
//
//  Block = 256 threads (8 waves). Block tile: 128 rows (M) x 64 cols (N).
//  Each wave: one 16-row tile x 4 N-tiles (16x64), 4 f32 accumulators.
//  B strips (64 N-rows x 128-wide K chunk, 16 KB each) are DOUBLE-BUFFERED in
//  LDS via CDNA5 async copies (global_load_async_to_lds_b128, ASYNCcnt):
//  chunk c+1 streams in while WMMAs consume chunk c. B fragments preloaded
//  4-wide per K-slice so ds_loads batch ahead of the WMMA burst. A fragments
//  come from global (each wave owns unique rows; L2-resident).
//
// Fragment layouts as loaded here (wave32, same contiguous K order for A and B):
//   A 16x32 f16 : lanes 0-15 -> M=lane, K = k0..k0+15 contiguous
//                 lanes16-31 -> M=lane-16, K = k0+16..k0+31
//   B 32x16 f16 : lanes 0-15 -> N=lane, K = k0..k0+15 contiguous
//                 lanes16-31 -> N=lane-16, K = k0+16..k0+31
//   C/D 16x16 f32: VGPR v, lanes 0-15 -> (M=v,   N=lane)
//                          lanes16-31 -> (M=v+8, N=lane-16)
// ---------------------------------------------------------------------------
template<int KDIM, bool GELU, bool OUT_F32>
__global__ __launch_bounds__(256)
void gemm_wmma(const _Float16* __restrict__ A,
               const _Float16* __restrict__ W,
               const float* __restrict__ bias,
               _Float16* __restrict__ outh,
               float* __restrict__ outf,
               int M, int N) {
  __shared__ _Float16 Bs[2][64 * 128];        // 2 x 16 KB double buffer

  const int nStrips = N >> 6;                  // N / 64
  const int bm = blockIdx.x / nStrips;         // 128-row block
  const int bn = blockIdx.x % nStrips;         // 64-col strip
  const int lane = threadIdx.x & 31;
  const int wave = threadIdx.x >> 5;
  const int row  = lane & 15;
  const int half = lane >> 4;

  const _Float16* aBase =
      A + (size_t)(bm * 128 + wave * 16 + row) * KDIM + half * 16;

  // async-stage one 64 x 128-half B chunk into Bs[buf]
  auto stage = [&](int kc, int buf) {
#pragma unroll
    for (int t = 0; t < 4; ++t) {
      const int idx = threadIdx.x + t * 256;   // 0..1023 b128 transfers
      const int r = idx >> 4;                  // B row 0..63
      const int c = (idx & 15) * 8;            // K half offset 0..120
      const unsigned long long g = (unsigned long long)
          (W + (size_t)(bn * 64 + r) * KDIM + kc + c);
      const unsigned loff = lds_offset(&Bs[buf][r * 128 + c]);
      asm volatile("global_load_async_to_lds_b128 %0, %1, off"
                   :: "v"(loff), "v"(g) : "memory");
    }
  };

  v8f acc[4];
#pragma unroll
  for (int t = 0; t < 4; ++t) acc[t] = (v8f){};

  constexpr int NCHUNK = KDIM / 128;
  stage(0, 0);
  for (int c = 0; c < NCHUNK; ++c) {
    asm volatile("s_wait_asynccnt 0" ::: "memory");
    __syncthreads();                            // whole strip visible in LDS
    if (c + 1 < NCHUNK) stage((c + 1) * 128, (c + 1) & 1);  // overlap next DMA

    const _Float16* bs = Bs[c & 1];
    const int kc = c * 128;
#pragma unroll
    for (int k0 = 0; k0 < 128; k0 += 32) {
      v8h alo = *(const v8h*)(aBase + kc + k0);        // K = +kb .. +7
      v8h ahi = *(const v8h*)(aBase + kc + k0 + 8);    // K = +8 .. +15
      v16h a = __builtin_shufflevector(alo, ahi,
                   0,1,2,3,4,5,6,7,8,9,10,11,12,13,14,15);
      // preload all 4 B fragments, then burst 4 WMMAs
      v16h b0 = *(const v16h*)&bs[(0 * 16 + row) * 128 + half * 16 + k0];
      v16h b1 = *(const v16h*)&bs[(1 * 16 + row) * 128 + half * 16 + k0];
      v16h b2 = *(const v16h*)&bs[(2 * 16 + row) * 128 + half * 16 + k0];
      v16h b3 = *(const v16h*)&bs[(3 * 16 + row) * 128 + half * 16 + k0];
      acc[0] = __builtin_amdgcn_wmma_f32_16x16x32_f16(
          false, a, false, b0, (short)0, acc[0], false, false);
      acc[1] = __builtin_amdgcn_wmma_f32_16x16x32_f16(
          false, a, false, b1, (short)0, acc[1], false, false);
      acc[2] = __builtin_amdgcn_wmma_f32_16x16x32_f16(
          false, a, false, b2, (short)0, acc[2], false, false);
      acc[3] = __builtin_amdgcn_wmma_f32_16x16x32_f16(
          false, a, false, b3, (short)0, acc[3], false, false);
    }
  }

  // ---- epilogue: bias (+ exact GELU), store 16x64 strip ----
  const int mrow0 = bm * 128 + wave * 16 + (lane >> 4) * 8;
#pragma unroll
  for (int t = 0; t < 4; ++t) {
    const int ncol = bn * 64 + t * 16 + (lane & 15);
    const float bv = bias[ncol];
#pragma unroll
    for (int v = 0; v < 8; ++v) {
      float x = acc[t][v] + bv;
      if (GELU) x = 0.5f * x * (1.0f + erff(x * 0.70710678118654752f));
      const size_t idx = (size_t)(mrow0 + v) * N + ncol;
      if (OUT_F32) outf[idx] = x;
      else         outh[idx] = (_Float16)x;
    }
  }
}

// ---------------------------------------------------------------------------
// Neighborhood attention (2D, 7x7 clipped window), online softmax.
// One thread per (pixel, head). qkv layout: [tok, 384] = [q(128)|k(128)|v(128)],
// channels head-major within each 128 (head*32 + d) — matches reference
// head split/concat. Output: f16 [tok, 128] ready as next GEMM's A.
// ---------------------------------------------------------------------------
__global__ void na2d_attn(const _Float16* __restrict__ qkv,
                          _Float16* __restrict__ o) {
  const int t = blockIdx.x * blockDim.x + threadIdx.x;
  if (t >= NTOK * NHEADS) return;
  const int head = t & (NHEADS - 1);
  const int p    = t >> 2;
  const int i = p / IMG_W, j = p % IMG_W;
  int sh = i - KWIN / 2; sh = sh < 0 ? 0 : sh; sh = sh > IMG_H - KWIN ? IMG_H - KWIN : sh;
  int sw = j - KWIN / 2; sw = sw < 0 ? 0 : sw; sw = sw > IMG_W - KWIN ? IMG_W - KWIN : sw;

  const _Float16* qp = qkv + (size_t)p * (3 * CDIM) + head * HDIM;
  float q[HDIM];
  const v8h* qv = (const v8h*)qp;
#pragma unroll
  for (int c = 0; c < HDIM / 8; ++c) {
    v8h tq = qv[c];
#pragma unroll
    for (int d = 0; d < 8; ++d) q[c * 8 + d] = (float)tq[d];
  }

  const float scale = 0.17677669529663687f;   // 1/sqrt(32)
  float m = -1e30f, l = 0.0f;
  float acc[HDIM];
#pragma unroll
  for (int d = 0; d < HDIM; ++d) acc[d] = 0.0f;

  for (int kh = 0; kh < KWIN; ++kh) {
    for (int kw = 0; kw < KWIN; ++kw) {
      const int np = (sh + kh) * IMG_W + (sw + kw);
      const _Float16* kp = qkv + (size_t)np * (3 * CDIM) + CDIM + head * HDIM;
      const v8h* kv = (const v8h*)kp;
      float s = 0.0f;
#pragma unroll
      for (int c = 0; c < HDIM / 8; ++c) {
        v8h tk = kv[c];
#pragma unroll
        for (int d = 0; d < 8; ++d) s += q[c * 8 + d] * (float)tk[d];
      }
      s *= scale;
      const float mn   = fmaxf(m, s);
      const float corr = __expf(m - mn);
      const float e    = __expf(s - mn);
      l = l * corr + e;
      const v8h* vv = (const v8h*)(kp + CDIM);   // v is 128 channels after k
#pragma unroll
      for (int c = 0; c < HDIM / 8; ++c) {
        v8h tv = vv[c];
#pragma unroll
        for (int d = 0; d < 8; ++d)
          acc[c * 8 + d] = acc[c * 8 + d] * corr + e * (float)tv[d];
      }
      m = mn;
    }
  }

  const float inv = 1.0f / l;
  _Float16* op = o + (size_t)p * CDIM + head * HDIM;
#pragma unroll
  for (int d = 0; d < HDIM; ++d) op[d] = (_Float16)(acc[d] * inv);
}

// ---------------------------------------------------------------------------
// Host-side orchestration
// ---------------------------------------------------------------------------
static inline int gemm_blocks(int M, int N) {
  return (M / 128) * (N / 64);                 // 128x64 block tiles
}

extern "C" void kernel_launch(void* const* d_in, const int* in_sizes, int n_in,
                              void* d_out, int out_size, void* d_ws, size_t ws_size,
                              hipStream_t stream) {
  const float* x      = (const float*)d_in[0];   // [1,96,96,128]
  const float* qkv_w  = (const float*)d_in[1];   // [384,128]
  const float* qkv_b  = (const float*)d_in[2];   // [384]
  const float* proj_w = (const float*)d_in[3];   // [128,128]
  const float* proj_b = (const float*)d_in[4];   // [128]
  const float* fc1_w  = (const float*)d_in[5];   // [512,128]
  const float* fc1_b  = (const float*)d_in[6];   // [512]
  const float* fc2_w  = (const float*)d_in[7];   // [128,512]
  const float* fc2_b  = (const float*)d_in[8];   // [128]
  float* out = (float*)d_out;                    // [9216,128] f32

  // Workspace layout (all offsets multiples of 256B)
  char* ws = (char*)d_ws;
  size_t off = 0;
  _Float16* xh    = (_Float16*)(ws + off); off += (size_t)NTOK * CDIM * 2;        // 2.25 MB
  _Float16* qkvwh = (_Float16*)(ws + off); off += (size_t)384 * 128 * 2;
  _Float16* projwh= (_Float16*)(ws + off); off += (size_t)128 * 128 * 2;
  _Float16* fc1wh = (_Float16*)(ws + off); off += (size_t)512 * 128 * 2;
  _Float16* fc2wh = (_Float16*)(ws + off); off += (size_t)128 * 512 * 2;
  _Float16* qkvh  = (_Float16*)(ws + off); off += (size_t)NTOK * 384 * 2;         // 6.75 MB
  _Float16* oh    = (_Float16*)(ws + off); off += (size_t)NTOK * CDIM * 2;
  _Float16* x0h   = (_Float16*)(ws + off); off += (size_t)NTOK * CDIM * 2;
  _Float16* h1h   = (_Float16*)(ws + off); off += (size_t)NTOK * 512 * 2;         // 9 MB

  // 1) f16 conversions
  {
    int n = NTOK * CDIM;
    cvt_f32_to_f16<<<(n + 1023) / 1024, 256, 0, stream>>>(x, xh, n);
    n = 384 * 128; cvt_f32_to_f16<<<(n + 255) / 256, 256, 0, stream>>>(qkv_w,  qkvwh,  n);
    n = 128 * 128; cvt_f32_to_f16<<<(n + 255) / 256, 256, 0, stream>>>(proj_w, projwh, n);
    n = 512 * 128; cvt_f32_to_f16<<<(n + 255) / 256, 256, 0, stream>>>(fc1_w,  fc1wh,  n);
    n = 128 * 512; cvt_f32_to_f16<<<(n + 255) / 256, 256, 0, stream>>>(fc2_w,  fc2wh,  n);
  }

  // 2) QKV projection: [9216,128] x [384,128]^T -> f16 [9216,384]
  gemm_wmma<128, false, false><<<gemm_blocks(NTOK, 384), 256, 0, stream>>>(
      xh, qkvwh, qkv_b, qkvh, nullptr, NTOK, 384);

  // 3) Neighborhood attention -> f16 [9216,128]
  {
    int n = NTOK * NHEADS;
    na2d_attn<<<(n + 255) / 256, 256, 0, stream>>>(qkvh, oh);
  }

  // 4) Output projection: -> f16 [9216,128]
  gemm_wmma<128, false, false><<<gemm_blocks(NTOK, 128), 256, 0, stream>>>(
      oh, projwh, proj_b, x0h, nullptr, NTOK, 128);

  // 5) FC1 + exact GELU: -> f16 [9216,512]
  gemm_wmma<128, true, false><<<gemm_blocks(NTOK, 512), 256, 0, stream>>>(
      x0h, fc1wh, fc1_b, h1h, nullptr, NTOK, 512);

  // 6) FC2: -> f32 [9216,128] into d_out
  gemm_wmma<512, false, true><<<gemm_blocks(NTOK, 128), 256, 0, stream>>>(
      h1h, fc2wh, fc2_b, nullptr, out, NTOK, 128);

  (void)in_sizes; (void)n_in; (void)out_size; (void)ws_size;
}
